// Finetuning_65000035057686
// MI455X (gfx1250) — matrix operand order repaired and checked by reading, not compiled, not run on hardware
//
#include <hip/hip_runtime.h>
#include <hip/hip_bf16.h>

#define NB 256
#define NA 64
#define NN 16384
#define NE 196608
#define NLE 393216
#define DD 256
#define NH 8
#define DHD 32
#define EDGE_D 76
#define LINE_D 30
#define PE_D 46
#define EMB_D 64
#define EKP 96      // EDGE_D padded to 32
#define LKP 32      // LINE_D padded to 32

typedef __attribute__((ext_vector_type(16))) __bf16 v16bf;
typedef __attribute__((ext_vector_type(8)))  __bf16 v8bf;
typedef __attribute__((ext_vector_type(8)))  float  v8f;

__device__ __forceinline__ float d_sp(float x){ return x > 20.f ? x : log1pf(expf(x)); }
__device__ __forceinline__ float d_sg(float x){ return 1.f / (1.f + expf(-x)); }
__device__ __forceinline__ __bf16 f2bf(float f){
  union { float f; unsigned u; } x; x.f = f;
  unsigned r = x.u + 0x7FFFu + ((x.u >> 16) & 1u);
  union { unsigned short s; __bf16 b; } y; y.s = (unsigned short)(r >> 16);
  return y.b;
}

// ---------------------------------------------------------------------------
// bf16 WMMA GEMM on pre-converted operands.
//   A  : bf16 [M, Kp] row-major (rows optionally gathered via arows)
//   WT : bf16 [N, Kp] row-major (weight pre-transposed)
//   C  : f32  [M, N] = actO( A@W + bias + resid )
// One 16x16 C tile per wave; inner loop = 4x b128 loads + 1 wmma.
// ---------------------------------------------------------------------------
__global__ void k_gemm_bf(const __bf16* __restrict__ A, const int* __restrict__ arows,
                          const __bf16* __restrict__ WT, const float* __restrict__ bias,
                          const float* __restrict__ resid, float* __restrict__ C,
                          int M, int Kp, int N, int actO)
{
  int tnc  = (N + 15) >> 4;
  int T    = ((M + 15) >> 4) * tnc;
  int tile = blockIdx.x * 4 + (threadIdx.x >> 5);
  if (tile >= T) return;
  int lane = threadIdx.x & 31;
  int tm = tile / tnc, tn = tile % tnc;
  int half = lane >> 4;
  int l15  = lane & 15;
  int row = tm * 16 + l15;
  int col = tn * 16 + l15;
  bool rok = row < M;
  bool cok = col < N;
  int rr = rok ? (arows ? arows[row] : row) : 0;   // clamp: bad rows/cols only feed
  int cc = cok ? col : 0;                          // outputs that are never stored
  const __bf16* Arow = A  + (long)rr * Kp;
  const __bf16* Brow = WT + (long)cc * Kp;
  v8f c = {0.f,0.f,0.f,0.f,0.f,0.f,0.f,0.f};
  for (int k0 = 0; k0 < Kp; k0 += 32) {
    v8bf alo = *(const v8bf*)(Arow + k0 + (half << 3));
    v8bf ahi = *(const v8bf*)(Arow + k0 + 16 + (half << 3));
    v8bf blo = *(const v8bf*)(Brow + k0 + (half << 3));
    v8bf bhi = *(const v8bf*)(Brow + k0 + 16 + (half << 3));
    v16bf a = __builtin_shufflevector(alo, ahi, 0,1,2,3,4,5,6,7,8,9,10,11,12,13,14,15);
    v16bf b = __builtin_shufflevector(blo, bhi, 0,1,2,3,4,5,6,7,8,9,10,11,12,13,14,15);
    c = __builtin_amdgcn_wmma_f32_16x16x32_bf16(false, a, false, b, (short)0, c, false, false);
  }
#pragma unroll
  for (int r = 0; r < 8; r++) {
    int m = tm * 16 + r + (half << 3);
    int n = tn * 16 + l15;
    if (m < M && n < N) {
      float val = c[r];
      if (bias)  val += bias[n];
      if (resid) val += resid[(long)m * N + n];
      if (actO)  val = d_sp(val);
      C[(long)m * N + n] = val;
    }
  }
}

// ---------------------------------------------------------------------------
// Operand preparation: f32 -> bf16 (K-padded), weight transpose, expansions
// ---------------------------------------------------------------------------
__global__ void k_a2bf(const float* __restrict__ A, __bf16* __restrict__ O,
                       long M, int K, int Kp, int act)
{
  long i = (long)blockIdx.x * blockDim.x + threadIdx.x;
  if (i >= M * (long)Kp) return;
  int  k = (int)(i % Kp);
  long m = i / Kp;
  float v = 0.f;
  if (k < K) { v = A[m * K + k]; if (act) v = d_sp(v); }
  O[i] = f2bf(v);
}

__global__ void k_wt(const float* __restrict__ W, __bf16* __restrict__ WT,
                     int K, int N, int Kp)
{
  int i = blockIdx.x * blockDim.x + threadIdx.x;
  if (i >= N * Kp) return;
  int n = i / Kp, k = i % Kp;
  WT[i] = f2bf(k < K ? W[(long)k * N + n] : 0.f);
}

__global__ void k_expand_nbr_bf(const float* __restrict__ sph, __bf16* __restrict__ out)
{
  int e = blockIdx.x * blockDim.x + threadIdx.x;
  if (e >= NE) return;
  float d0 = sph[e*3+0], d1 = sph[e*3+1], d2 = sph[e*3+2];
  __bf16* o = out + (long)e * EKP;
  for (int k = 0; k < 41; k++) { float t = d0 - 0.2f*k;           o[k]    = f2bf(expf(-t*t*25.0f)); }
  for (int k = 0; k < 17; k++) { float t = d1 - 0.2f*k;           o[41+k] = f2bf(expf(-t*t*25.0f)); }
  for (int k = 0; k < 17; k++) { float t = d2 - (-3.2f + 0.4f*k); o[58+k] = f2bf(expf(-t*t*6.25f)); }
  o[75] = f2bf(d0 > 8.0f ? 1.0f : 0.0f);
  for (int k = 76; k < EKP; k++) o[k] = f2bf(0.f);
}

__global__ void k_expand_line_bf(const float* __restrict__ d, __bf16* __restrict__ out)
{
  int e = blockIdx.x * blockDim.x + threadIdx.x;
  if (e >= NLE) return;
  float dv = d[e];
  __bf16* o = out + (long)e * LKP;
  for (int k = 0; k < 30; k++) { float t = dv - (-1.4f + 0.1f*k); o[k] = f2bf(expf(-t*t*100.0f)); }
  o[30] = f2bf(0.f); o[31] = f2bf(0.f);
}

// ---------------------------------------------------------------------------
// CGConv fused message + aggregate (block per graph-edge, LDS-staged e-row)
// ---------------------------------------------------------------------------
__global__ void k_msg(const int* __restrict__ src, const int* __restrict__ dst,
                      const float* __restrict__ XFi, const float* __restrict__ XFj,
                      const float* __restrict__ XSi, const float* __restrict__ XSj,
                      const float* __restrict__ ef,  const float* __restrict__ wfe,
                      const float* __restrict__ wse, float* __restrict__ agg,
                      int C, int ED)
{
  extern __shared__ float sh[];
  int e = blockIdx.x;
  int s = src[e], d = dst[e];
  for (int t = threadIdx.x; t < ED; t += blockDim.x) sh[t] = ef[(long)e * ED + t];
  __syncthreads();
  for (int c = threadIdx.x; c < C; c += blockDim.x) {
    float ff = XFi[(long)d * C + c] + XFj[(long)s * C + c];
    float ss = XSi[(long)d * C + c] + XSj[(long)s * C + c];
    for (int t = 0; t < ED; t++) {
      float ev = sh[t];
      ff += ev * wfe[(long)t * C + c];
      ss += ev * wse[(long)t * C + c];
    }
    atomicAdd(&agg[(long)d * C + c], d_sg(ff) * d_sp(ss));
  }
}

__global__ void k_bn_stats(const float* __restrict__ agg, int n, int C, float* __restrict__ mv)
{
  __shared__ float ss[256], sq[256];
  int c = blockIdx.x, tid = threadIdx.x;
  float s = 0.f, s2 = 0.f;
  for (int i = tid; i < n; i += 256) { float v = agg[(long)i * C + c]; s += v; s2 += v*v; }
  ss[tid] = s; sq[tid] = s2;
  __syncthreads();
  for (int st = 128; st > 0; st >>= 1) {
    if (tid < st) { ss[tid] += ss[tid+st]; sq[tid] += sq[tid+st]; }
    __syncthreads();
  }
  if (tid == 0) { float mu = ss[0] / n; mv[c] = mu; mv[C + c] = sq[0] / n - mu*mu; }
}

__global__ void k_bn_apply(const float* __restrict__ agg, const float* __restrict__ mv,
                           const float* __restrict__ g, const float* __restrict__ b,
                           float* __restrict__ x, long total, int C)
{
  long i = (long)blockIdx.x * blockDim.x + threadIdx.x;
  if (i >= total) return;
  int c = (int)(i % C);
  x[i] = g[c] * (agg[i] - mv[c]) * rsqrtf(mv[C + c] + 1e-5f) + b[c] + x[i];
}

__global__ void k_layernorm(const float* __restrict__ x, const float* __restrict__ g,
                            const float* __restrict__ b, float* __restrict__ y, int n, int C)
{
  int i = blockIdx.x * blockDim.x + threadIdx.x;
  if (i >= n) return;
  const float* row = x + (long)i * C;
  float mu = 0.f;
  for (int c = 0; c < C; c++) mu += row[c];
  mu /= C;
  float var = 0.f;
  for (int c = 0; c < C; c++) { float d = row[c] - mu; var += d*d; }
  var /= C;
  float inv = rsqrtf(var + 1e-5f);
  for (int c = 0; c < C; c++) y[(long)i * C + c] = g[c] * (row[c] - mu) * inv + b[c];
}

// ---------------------------------------------------------------------------
// Transformer: WMMA scores (bf16 q/k), bias scatter, softmax, WMMA attn@V
// ---------------------------------------------------------------------------
__global__ void k_attn_scores(const __bf16* __restrict__ q, const __bf16* __restrict__ kmat,
                              float* __restrict__ scores, float scale)
{
  int bh = blockIdx.x;                 // b*NH + h
  int b = bh >> 3, h = bh & 7;
  int wave = threadIdx.x >> 5, lane = threadIdx.x & 31;
  int half = lane >> 4, l15 = lane & 15;
  int mrow = wave * 16 + l15;
  const __bf16* qrow = q + (long)(b * NA + mrow) * DD + h * DHD;
  v8bf alo = *(const v8bf*)(qrow + (half << 3));
  v8bf ahi = *(const v8bf*)(qrow + 16 + (half << 3));
  v16bf a = __builtin_shufflevector(alo, ahi, 0,1,2,3,4,5,6,7,8,9,10,11,12,13,14,15);
  for (int tn = 0; tn < 4; tn++) {
    int col = tn * 16 + l15;
    const __bf16* krow = kmat + (long)(b * NA + col) * DD + h * DHD;
    v8bf blo = *(const v8bf*)(krow + (half << 3));
    v8bf bhi = *(const v8bf*)(krow + 16 + (half << 3));
    v16bf bb = __builtin_shufflevector(blo, bhi, 0,1,2,3,4,5,6,7,8,9,10,11,12,13,14,15);
    v8f c = {0.f,0.f,0.f,0.f,0.f,0.f,0.f,0.f};
    c = __builtin_amdgcn_wmma_f32_16x16x32_bf16(false, a, false, bb, (short)0, c, false, false);
#pragma unroll
    for (int r = 0; r < 8; r++) {
      int i = wave * 16 + r + (half << 3);
      scores[((long)bh * NA + i) * NA + col] = c[r] * scale;
    }
  }
}

__global__ void k_scatter_bias(const int* __restrict__ src, const int* __restrict__ dst,
                               const float* __restrict__ bias, float* __restrict__ scores)
{
  int e = blockIdx.x * blockDim.x + threadIdx.x;
  if (e >= NE) return;
  int d = dst[e], s = src[e];
  int bb = d >> 6, ii = d & 63, jj = s & 63;
  for (int h = 0; h < NH; h++)
    atomicAdd(&scores[(((long)(bb * NH + h)) * NA + ii) * NA + jj], bias[(long)e * NH + h]);
}

// softmax over 64; writes bf16 probabilities for the WMMA attn@V stage
__global__ void k_softmax64(float* __restrict__ s, __bf16* __restrict__ p, int rows)
{
  int r = blockIdx.x * blockDim.x + threadIdx.x;
  if (r >= rows) return;
  float* row = s + (long)r * NA;
  __bf16* pr = p + (long)r * NA;
  float m = row[0];
  for (int j = 1; j < NA; j++) m = fmaxf(m, row[j]);
  float sum = 0.f;
  for (int j = 0; j < NA; j++) { float e = expf(row[j] - m); row[j] = e; sum += e; }
  float inv = 1.f / sum;
  for (int j = 0; j < NA; j++) pr[j] = f2bf(row[j] * inv);
}

// vT[bh][d][j] = v[b*64+j][h*32+d]  (bf16), so attn@V B-fragments are contiguous
__global__ void k_vT(const float* __restrict__ v, __bf16* __restrict__ vt)
{
  long i = (long)blockIdx.x * blockDim.x + threadIdx.x;
  if (i >= (long)NB * NH * DHD * NA) return;
  int  j  = (int)(i & 63);
  long t  = i >> 6;
  int  d  = (int)(t & 31);
  long bh = t >> 5;
  int  b  = (int)(bh >> 3), h = (int)(bh & 7);
  vt[i] = f2bf(v[(long)(b * NA + j) * DD + h * DHD + d]);
}

__global__ void k_attn_v(const __bf16* __restrict__ probs, const __bf16* __restrict__ vt,
                         float* __restrict__ o)
{
  int bh = blockIdx.x;
  int b = bh >> 3, h = bh & 7;
  int wave = threadIdx.x >> 5, lane = threadIdx.x & 31;
  int half = lane >> 4, l15 = lane & 15;
  int mrow = wave * 16 + l15;
  const __bf16* arow = probs + ((long)bh * NA + mrow) * NA;
  for (int tn = 0; tn < 2; tn++) {
    int col = tn * 16 + l15;
    const __bf16* brow = vt + ((long)bh * DHD + col) * NA;
    v8f c = {0.f,0.f,0.f,0.f,0.f,0.f,0.f,0.f};
    for (int k0 = 0; k0 < NA; k0 += 32) {
      v8bf alo = *(const v8bf*)(arow + k0 + (half << 3));
      v8bf ahi = *(const v8bf*)(arow + k0 + 16 + (half << 3));
      v8bf blo = *(const v8bf*)(brow + k0 + (half << 3));
      v8bf bhi = *(const v8bf*)(brow + k0 + 16 + (half << 3));
      v16bf a = __builtin_shufflevector(alo, ahi, 0,1,2,3,4,5,6,7,8,9,10,11,12,13,14,15);
      v16bf bb = __builtin_shufflevector(blo, bhi, 0,1,2,3,4,5,6,7,8,9,10,11,12,13,14,15);
      c = __builtin_amdgcn_wmma_f32_16x16x32_bf16(false, a, false, bb, (short)0, c, false, false);
    }
#pragma unroll
    for (int r = 0; r < 8; r++) {
      int i = wave * 16 + r + (half << 3);
      o[(long)(b * NA + i) * DD + h * DHD + col] = c[r];
    }
  }
}

// ---------------------------------------------------------------------------
// Misc elementwise
// ---------------------------------------------------------------------------
__global__ void k_addv(float* __restrict__ a, const float* __restrict__ b, long n)
{
  long i = (long)blockIdx.x * blockDim.x + threadIdx.x;
  if (i < n) a[i] += b[i];
}

__global__ void k_pool(const float* __restrict__ atom, float* __restrict__ out)
{
  int i = blockIdx.x * blockDim.x + threadIdx.x;
  if (i >= NB * DD) return;
  int b = i / DD, c = i % DD;
  float s = 0.f;
  for (int a = 0; a < NA; a++) s += atom[(long)(b * NA + a) * DD + c];
  out[i] = d_sp(s * (1.0f / NA));
}

__global__ void k_fcout(const float* __restrict__ t, const float* __restrict__ w,
                        const float* __restrict__ b, float* __restrict__ out)
{
  int i = blockIdx.x * blockDim.x + threadIdx.x;
  if (i >= NB) return;
  float acc = b[0];
  for (int k = 0; k < DD; k++) acc += d_sp(t[(long)i * DD + k]) * w[k];
  out[i] = acc;
}

// ---------------------------------------------------------------------------
extern "C" void kernel_launch(void* const* d_in, const int* in_sizes, int n_in,
                              void* d_out, int out_size, void* d_ws, size_t ws_size,
                              hipStream_t stream)
{
  const int*   atom_ids = (const int*)  d_in[0];
  const float* spherical= (const float*)d_in[1];
  const float* pe       = (const float*)d_in[2];
  const float* line_sc  = (const float*)d_in[3];
  const int*   eidx     = (const int*)  d_in[4];
  const int*   lidx     = (const int*)  d_in[5];
  auto Pf = [&](int i) { return (const float*)d_in[6 + i]; };
  // param leaf map (insertion-order flatten):
  // 0 emb | 1,2 emb2hid | 3,4 edge2nbr | 5,6 line2line | 7,8 pe2hid
  // 9..38 line_convs[3]x{wf_i,wf_j,wf_e,bf,ws_i,ws_j,ws_e,bs,bn_g,bn_b}
  // 39..68 atom_convs[3] | 69,70 lnn_g/b | 71..80 gt1 | 81..90 gt2
  // 91,92 conv_to_fc | 93..96 fcs | 97,98 fc_out
  const int* esrc = eidx;  const int* edst = eidx + NE;
  const int* lsrc = lidx;  const int* ldst = lidx + NLE;

  char* basep = (char*)d_ws;
  size_t off = 0;
  auto takeB = [&](size_t bytes) { void* p = basep + off; off += (bytes + 255) & ~(size_t)255; return p; };
  auto takeF = [&](size_t n)     { return (float*)takeB(n * sizeof(float)); };

  float* nbr   = takeF((size_t)NE * EDGE_D);
  float* line  = takeF((size_t)NLE * LINE_D);
  float* atomF = takeF((size_t)NN * DD);
  float* peh   = takeF((size_t)NN * DD);
  float* lnnbr = takeF((size_t)NE * EDGE_D);
  float* stats = takeF(512);
  size_t S     = (size_t)NE * EDGE_D;          // f32 scratch slot
  float* pool  = takeF(5 * S);
  __bf16* bf1  = (__bf16*)takeB((size_t)NE * EKP * 2);            // 18.9M elems
  __bf16* bf2  = (__bf16*)takeB((size_t)NB * NH * NA * NA * 2);   // 8.4M elems
  __bf16* bf3  = (__bf16*)takeB((size_t)NB * NH * NA * NA * 2);
  __bf16* embb = (__bf16*)takeB((size_t)100 * EMB_D * 2);

  auto gemm = [&](const __bf16* A, const int* arows, const __bf16* WT, const float* bias,
                  const float* resid, float* C, int M, int Kp, int N, int actO) {
    int T = ((M + 15) / 16) * ((N + 15) / 16);
    k_gemm_bf<<<dim3((T + 3) / 4), dim3(128), 0, stream>>>(A, arows, WT, bias, resid, C,
                                                           M, Kp, N, actO);
  };
  auto cvt = [&](const float* A, __bf16* O, long M, int K, int Kp, int act) {
    long tot = M * (long)Kp;
    k_a2bf<<<dim3((unsigned)((tot + 255) / 256)), dim3(256), 0, stream>>>(A, O, M, K, Kp, act);
  };
  auto wt = [&](int leaf, int K, int N, int Kp) {
    __bf16* p = (__bf16*)takeB((size_t)N * Kp * 2);
    int tot = N * Kp;
    k_wt<<<dim3((tot + 255) / 256), dim3(256), 0, stream>>>(Pf(leaf), p, K, N, Kp);
    return (const __bf16*)p;
  };

  // ---- weight preparation (transpose + bf16, K padded) ---------------------
  const __bf16* Wt_e2n = wt(3, EDGE_D, EDGE_D, EKP);
  const __bf16* Wt_l2l = wt(5, LINE_D, LINE_D, LKP);
  const __bf16* Wt_emb = wt(1, EMB_D, DD, EMB_D);
  const __bf16* Wt_pe  = wt(7, PE_D, DD, 64);
  const __bf16* Wt_lc[3][4];
  const __bf16* Wt_ac[3][4];
  for (int i = 0; i < 3; i++) {
    int pL = 9 + 10*i, pA = 39 + 10*i;
    int sel[4] = {0, 1, 4, 5};                      // wf_i, wf_j, ws_i, ws_j
    for (int j = 0; j < 4; j++) {
      Wt_lc[i][j] = wt(pL + sel[j], EDGE_D, EDGE_D, EKP);
      Wt_ac[i][j] = wt(pA + sel[j], DD, DD, DD);
    }
  }
  const __bf16* Wt_gt[2][5];
  for (int t = 0; t < 2; t++) {
    int g = 71 + 10*t;
    Wt_gt[t][0] = wt(g+0, DD, DD, DD);              // q
    Wt_gt[t][1] = wt(g+2, DD, DD, DD);              // k
    Wt_gt[t][2] = wt(g+4, DD, DD, DD);              // v
    Wt_gt[t][3] = wt(g+6, DD, DD, DD);              // o
    Wt_gt[t][4] = wt(g+8, EDGE_D, NH, EKP);         // e bias
  }
  const __bf16* Wt_hd[3] = { wt(91, DD, DD, DD), wt(93, DD, DD, DD), wt(95, DD, DD, DD) };

  // ---- input featurization -------------------------------------------------
  k_expand_nbr_bf<<<dim3((NE + 255) / 256), dim3(256), 0, stream>>>(spherical, bf1);
  gemm(bf1, nullptr, Wt_e2n, Pf(4), nullptr, nbr, NE, EKP, EDGE_D, 0);
  k_expand_line_bf<<<dim3((NLE + 255) / 256), dim3(256), 0, stream>>>(line_sc, bf1);
  gemm(bf1, nullptr, Wt_l2l, Pf(6), nullptr, line, NLE, LKP, LINE_D, 0);
  cvt(Pf(0), embb, 100, EMB_D, EMB_D, 0);
  gemm(embb, atom_ids, Wt_emb, Pf(2), nullptr, atomF, NN, EMB_D, DD, 0);
  cvt(pe, bf2, NN, PE_D, 64, 0);
  gemm(bf2, nullptr, Wt_pe, Pf(8), nullptr, peh, NN, 64, DD, 0);

  // ---- message-passing convolutions ---------------------------------------
  float* XFi = pool;  float* XFj = pool + S;  float* XSi = pool + 2*S;
  float* XSj = pool + 3*S;  float* agg = pool + 4*S;
  for (int i = 0; i < 3; i++) {
    int pL = 9 + 10*i, pA = 39 + 10*i;
    // line-graph CGConv on nbr (nodes = E edges)
    cvt(nbr, bf1, NE, EDGE_D, EKP, 0);
    gemm(bf1, nullptr, Wt_lc[i][0], Pf(pL+3), nullptr, XFi, NE, EKP, EDGE_D, 0); // wf_i+bf
    gemm(bf1, nullptr, Wt_lc[i][1], nullptr,  nullptr, XFj, NE, EKP, EDGE_D, 0); // wf_j
    gemm(bf1, nullptr, Wt_lc[i][2], Pf(pL+7), nullptr, XSi, NE, EKP, EDGE_D, 0); // ws_i+bs
    gemm(bf1, nullptr, Wt_lc[i][3], nullptr,  nullptr, XSj, NE, EKP, EDGE_D, 0); // ws_j
    hipMemsetAsync(agg, 0, (size_t)NE * EDGE_D * sizeof(float), stream);
    k_msg<<<dim3(NLE), dim3(128), LINE_D * sizeof(float), stream>>>(
        lsrc, ldst, XFi, XFj, XSi, XSj, line, Pf(pL+2), Pf(pL+6), agg, EDGE_D, LINE_D);
    k_bn_stats<<<dim3(EDGE_D), dim3(256), 0, stream>>>(agg, NE, EDGE_D, stats);
    k_bn_apply<<<dim3((unsigned)(((long)NE*EDGE_D + 255) / 256)), dim3(256), 0, stream>>>(
        agg, stats, Pf(pL+8), Pf(pL+9), nbr, (long)NE*EDGE_D, EDGE_D);
    // atom-graph CGConv, edge feats = LayerNorm(nbr)
    k_layernorm<<<dim3((NE + 255) / 256), dim3(256), 0, stream>>>(nbr, Pf(69), Pf(70),
                                                                  lnnbr, NE, EDGE_D);
    cvt(atomF, bf2, NN, DD, DD, 0);
    gemm(bf2, nullptr, Wt_ac[i][0], Pf(pA+3), nullptr, XFi, NN, DD, DD, 0);
    gemm(bf2, nullptr, Wt_ac[i][1], nullptr,  nullptr, XFj, NN, DD, DD, 0);
    gemm(bf2, nullptr, Wt_ac[i][2], Pf(pA+7), nullptr, XSi, NN, DD, DD, 0);
    gemm(bf2, nullptr, Wt_ac[i][3], nullptr,  nullptr, XSj, NN, DD, DD, 0);
    hipMemsetAsync(agg, 0, (size_t)NN * DD * sizeof(float), stream);
    k_msg<<<dim3(NE), dim3(256), EDGE_D * sizeof(float), stream>>>(
        esrc, edst, XFi, XFj, XSi, XSj, lnnbr, Pf(pA+2), Pf(pA+6), agg, DD, EDGE_D);
    k_bn_stats<<<dim3(DD), dim3(256), 0, stream>>>(agg, NN, DD, stats);
    k_bn_apply<<<dim3((unsigned)(((long)NN*DD + 255) / 256)), dim3(256), 0, stream>>>(
        agg, stats, Pf(pA+8), Pf(pA+9), atomF, (long)NN*DD, DD);
  }

  // ---- positional encoding residual ---------------------------------------
  k_addv<<<dim3((NN * DD + 255) / 256), dim3(256), 0, stream>>>(atomF, peh, (long)NN * DD);

  // ---- global transformers -------------------------------------------------
  cvt(nbr, bf1, NE, EDGE_D, EKP, 0);               // nbr_bf, reused by both gts
  float* qb = pool;  float* kb = pool + S;  float* vb = pool + 2*S;  float* ob = pool + 3*S;
  float* scores = pool + 4*S;
  float* bias   = takeF((size_t)NE * NH);
  const float inv_sqrt_dh = 0.17677669529663688f;  // 1/sqrt(32)
  for (int t = 0; t < 2; t++) {
    int g = 71 + 10*t;
    cvt(atomF, bf2, NN, DD, DD, 0);                // atom_bf
    gemm(bf2, nullptr, Wt_gt[t][0], Pf(g+1), nullptr, qb, NN, DD, DD, 0);
    gemm(bf2, nullptr, Wt_gt[t][1], Pf(g+3), nullptr, kb, NN, DD, DD, 0);
    gemm(bf2, nullptr, Wt_gt[t][2], Pf(g+5), nullptr, vb, NN, DD, DD, 0);
    gemm(bf1, nullptr, Wt_gt[t][4], Pf(g+9), nullptr, bias, NE, EKP, NH, 0);
    cvt(qb, bf2, NN, DD, DD, 0);                   // q_bf (atom_bf dead)
    cvt(kb, bf3, NN, DD, DD, 0);                   // k_bf
    k_attn_scores<<<dim3(NB * NH), dim3(128), 0, stream>>>(bf2, bf3, scores, inv_sqrt_dh);
    k_scatter_bias<<<dim3((NE + 255) / 256), dim3(256), 0, stream>>>(esrc, edst, bias, scores);
    k_softmax64<<<dim3((NB * NH * NA + 255) / 256), dim3(256), 0, stream>>>(
        scores, bf2, NB * NH * NA);                // probs_bf (q_bf dead)
    k_vT<<<dim3((unsigned)(((long)NB*NH*DHD*NA + 255) / 256)), dim3(256), 0, stream>>>(vb, bf3);
    k_attn_v<<<dim3(NB * NH), dim3(128), 0, stream>>>(bf2, bf3, ob);
    cvt(ob, bf3, NN, DD, DD, 0);                   // ob_bf
    gemm(bf3, nullptr, Wt_gt[t][3], Pf(g+7), atomF, atomF, NN, DD, DD, 1); // +resid, softplus
  }

  // ---- pooled MLP head -----------------------------------------------------
  float* c0 = pool;  float* c1 = pool + S;  float* c2 = pool + 2*S;
  k_pool<<<dim3((NB * DD + 255) / 256), dim3(256), 0, stream>>>(atomF, c0);
  cvt(c0, bf2, NB, DD, DD, 0);
  gemm(bf2, nullptr, Wt_hd[0], Pf(92), nullptr, c1, NB, DD, DD, 0);
  cvt(c1, bf2, NB, DD, DD, 1);                     // softplus pre-activation
  gemm(bf2, nullptr, Wt_hd[1], Pf(94), nullptr, c2, NB, DD, DD, 0);
  cvt(c2, bf2, NB, DD, DD, 1);
  gemm(bf2, nullptr, Wt_hd[2], Pf(96), nullptr, c0, NB, DD, DD, 0);
  k_fcout<<<dim3(1), dim3(256), 0, stream>>>(c0, Pf(97), Pf(98), (float*)d_out);
}
